// ASKLoss_89532888252789
// MI455X (gfx1250) — compile-verified, matrix operand order checked by reading, not run on hardware
//
#include <hip/hip_runtime.h>
#include <hip/hip_bf16.h>

typedef _Float16 v16h __attribute__((ext_vector_type(16)));
typedef _Float16 h8   __attribute__((ext_vector_type(8)));
typedef float    v8f  __attribute__((ext_vector_type(8)));

#define EPSF 1e-6f

// ---------------------------------------------------------------------------
// Phase 1: f32 -> f16 conversion + exact row squared-norms (one wave32 / row)
// ---------------------------------------------------------------------------
__global__ __launch_bounds__(256) void snn_prep(
    const float* __restrict__ x, const float* __restrict__ xref,
    _Float16* __restrict__ xh, _Float16* __restrict__ rh,
    float* __restrict__ x2, float* __restrict__ r2,
    int M, int N, int D) {
  const int wave = threadIdx.x >> 5;
  const int lane = threadIdx.x & 31;
  const int row  = blockIdx.x * 8 + wave;

  const float* src;
  _Float16*    dst;
  float*       nrm;
  if (row < M) {
    src = x + (size_t)row * D;  dst = xh + (size_t)row * D;  nrm = x2 + row;
  } else {
    int r = row - M;
    if (r >= N) return;
    src = xref + (size_t)r * D; dst = rh + (size_t)r * D;    nrm = r2 + r;
  }

  struct alignas(8) H4 { _Float16 a, b, c, d; };
  const float4* s4 = reinterpret_cast<const float4*>(src);
  H4*           d4 = reinterpret_cast<H4*>(dst);

  float acc = 0.f;
  const int nch = D >> 7;               // D / (32 lanes * 4 floats)
  for (int j = 0; j < nch; ++j) {
    float4 v = s4[lane * nch + j];
    acc += v.x * v.x + v.y * v.y + v.z * v.z + v.w * v.w;
    H4 o; o.a = (_Float16)v.x; o.b = (_Float16)v.y;
          o.c = (_Float16)v.z; o.d = (_Float16)v.w;
    d4[lane * nch + j] = o;
  }
  for (int off = 16; off > 0; off >>= 1) acc += __shfl_xor(acc, off, 32);
  if (lane == 0) *nrm = acc;
}

// ---------------------------------------------------------------------------
// Phase 2: fused GEMM (WMMA f16->f32) + sqrt/exp + per-row online sums
// One workgroup = 16 query rows; 8 waves split the 32768 reference columns,
// 4 N-tiles per wave iteration to amortize A-fragment LDS reads 4x.
// ---------------------------------------------------------------------------
__global__ __launch_bounds__(256) void snn_main(
    const _Float16* __restrict__ xh, const _Float16* __restrict__ rh,
    const float* __restrict__ x2, const float* __restrict__ r2,
    const int* __restrict__ y, const int* __restrict__ yref,
    float* __restrict__ lossv, int N, int D) {
  __shared__ _Float16 As[16 * 512];     // A panel: 16 rows x K=512 (f16)
  __shared__ float x2s[16];
  __shared__ int   ys[16];
  __shared__ float ptot[128];           // [wave][row] partial denominators
  __shared__ float ptgt[128];           // [wave][row] partial target sums

  const int tid     = threadIdx.x;
  const int wave    = tid >> 5;
  const int lane    = tid & 31;
  const int rowbase = blockIdx.x * 16;

  // Stage A panel global->LDS in 16B chunks
  {
    const uint4* src = reinterpret_cast<const uint4*>(xh + (size_t)rowbase * D);
    uint4*       dst = reinterpret_cast<uint4*>(As);
    const int total = (16 * D) >> 3;    // # of 16-byte chunks
    for (int i = tid; i < total; i += 256) dst[i] = src[i];
    if (tid < 16) { x2s[tid] = x2[rowbase + tid]; ys[tid] = y[rowbase + tid]; }
  }
  __syncthreads();

  // WMMA 16-bit fragment layout (ISA 7.12.2): lane L holds row/col L&15;
  // halves 0-7 -> K = c0..c0+7, halves 8-15 -> K = c0+16..c0+23, c0 = (L<16?0:8)
  const int rl = lane & 15;             // A row / B column within tile
  const int hs = lane >> 4;
  const int c0 = hs * 8;                // K sub-offset
  const int m0 = hs * 8;                // C/D row offset (lanes 16-31 hold M=8..15)

  float x2r[8]; int ymr[8];
  #pragma unroll
  for (int rI = 0; rI < 8; ++rI) { x2r[rI] = x2s[m0 + rI]; ymr[rI] = ys[m0 + rI]; }

  float stot[8], stgt[8];
  #pragma unroll
  for (int rI = 0; rI < 8; ++rI) { stot[rI] = 0.f; stgt[rI] = 0.f; }

  union AF { v16h v; h8 h[2]; };

  const int NT = N >> 4;                // 16-column tiles
  const int KC = D >> 5;                // 32-wide K chunks

  const _Float16* abase = As + rl * D + c0;

  // Epilogue helper: dist = sqrt(|x|^2 - 2 x.r + |r|^2); accumulate exp(-d)
  auto epilogue = [&](int t, const v8f& cacc) {
    const int n = t * 16 + rl;
    const float rc = r2[n]; const int cls = yref[n];
    #pragma unroll
    for (int rI = 0; rI < 8; ++rI) {
      float d2 = x2r[rI] + rc - 2.f * cacc[rI];
      float dd = sqrtf(fmaxf(d2, 0.f));
      float e  = __expf(-dd);
      stot[rI] += e;
      stgt[rI] += (cls == ymr[rI]) ? e : 0.f;
    }
  };

  for (int t0 = wave * 4; t0 < NT; t0 += 32) {
    v8f cacc0 = {}; v8f cacc1 = {}; v8f cacc2 = {}; v8f cacc3 = {};
    const _Float16* b0base = rh + ((size_t)((t0 + 0) * 16 + rl)) * D + c0;
    const _Float16* b1base = rh + ((size_t)((t0 + 1) * 16 + rl)) * D + c0;
    const _Float16* b2base = rh + ((size_t)((t0 + 2) * 16 + rl)) * D + c0;
    const _Float16* b3base = rh + ((size_t)((t0 + 3) * 16 + rl)) * D + c0;

    for (int k = 0; k < KC; ++k) {
      const int ko = k * 32;
      AF a, b0, b1, b2, b3;
      a.h[0]  = *reinterpret_cast<const h8*>(abase  + ko);
      a.h[1]  = *reinterpret_cast<const h8*>(abase  + ko + 16);
      b0.h[0] = *reinterpret_cast<const h8*>(b0base + ko);
      b0.h[1] = *reinterpret_cast<const h8*>(b0base + ko + 16);
      b1.h[0] = *reinterpret_cast<const h8*>(b1base + ko);
      b1.h[1] = *reinterpret_cast<const h8*>(b1base + ko + 16);
      b2.h[0] = *reinterpret_cast<const h8*>(b2base + ko);
      b2.h[1] = *reinterpret_cast<const h8*>(b2base + ko + 16);
      b3.h[0] = *reinterpret_cast<const h8*>(b3base + ko);
      b3.h[1] = *reinterpret_cast<const h8*>(b3base + ko + 16);
      cacc0 = __builtin_amdgcn_wmma_f32_16x16x32_f16(
                  false, a.v, false, b0.v, (short)0, cacc0, false, false);
      cacc1 = __builtin_amdgcn_wmma_f32_16x16x32_f16(
                  false, a.v, false, b1.v, (short)0, cacc1, false, false);
      cacc2 = __builtin_amdgcn_wmma_f32_16x16x32_f16(
                  false, a.v, false, b2.v, (short)0, cacc2, false, false);
      cacc3 = __builtin_amdgcn_wmma_f32_16x16x32_f16(
                  false, a.v, false, b3.v, (short)0, cacc3, false, false);
    }

    epilogue(t0 + 0, cacc0);
    epilogue(t0 + 1, cacc1);
    epilogue(t0 + 2, cacc2);
    epilogue(t0 + 3, cacc3);
  }

  // Deterministic reduction across the 16 lanes within each half (fixed order)
  #pragma unroll
  for (int rI = 0; rI < 8; ++rI) {
    float a = stot[rI], b = stgt[rI];
    for (int off = 8; off > 0; off >>= 1) {
      a += __shfl_xor(a, off, 32);
      b += __shfl_xor(b, off, 32);
    }
    stot[rI] = a; stgt[rI] = b;
  }
  if (rl == 0) {
    #pragma unroll
    for (int rI = 0; rI < 8; ++rI) {
      ptot[wave * 16 + m0 + rI] = stot[rI];
      ptgt[wave * 16 + m0 + rI] = stgt[rI];
    }
  }
  __syncthreads();

  if (tid < 16) {
    float tot = 0.f, tgt = 0.f;
    for (int w = 0; w < 8; ++w) {       // fixed order -> deterministic
      tot += ptot[w * 16 + tid];
      tgt += ptgt[w * 16 + tid];
    }
    lossv[rowbase + tid] = -__logf(tgt / tot + EPSF);
  }
}

// ---------------------------------------------------------------------------
// Phase 3: deterministic single-block mean reduction
// ---------------------------------------------------------------------------
__global__ __launch_bounds__(256) void snn_reduce(
    const float* __restrict__ lossv, float* __restrict__ out, int M) {
  __shared__ float sm[256];
  const int tid = threadIdx.x;
  float s = 0.f;
  for (int i = tid; i < M; i += 256) s += lossv[i];
  sm[tid] = s;
  __syncthreads();
  for (int off = 128; off > 0; off >>= 1) {
    if (tid < off) sm[tid] += sm[tid + off];
    __syncthreads();
  }
  if (tid == 0) out[0] = sm[0] / (float)M;
}

// ---------------------------------------------------------------------------
extern "C" void kernel_launch(void* const* d_in, const int* in_sizes, int n_in,
                              void* d_out, int out_size, void* d_ws, size_t ws_size,
                              hipStream_t stream) {
  const float* x    = (const float*)d_in[0];
  const float* xref = (const float*)d_in[1];
  const int*   y    = (const int*)d_in[2];
  const int*   yref = (const int*)d_in[3];

  const int M = in_sizes[2];            // 4096
  const int N = in_sizes[3];            // 32768
  const int D = in_sizes[0] / M;        // 512

  // Workspace layout
  char* ws = (char*)d_ws;
  size_t off = 0;
  _Float16* xh = (_Float16*)(ws + off); off += (size_t)M * D * sizeof(_Float16);
  _Float16* rh = (_Float16*)(ws + off); off += (size_t)N * D * sizeof(_Float16);
  float* x2    = (float*)(ws + off);    off += (size_t)M * sizeof(float);
  float* r2    = (float*)(ws + off);    off += (size_t)N * sizeof(float);
  float* lossv = (float*)(ws + off);    off += (size_t)M * sizeof(float);
  (void)ws_size; (void)n_in; (void)out_size;

  const int prep_blocks = (M + N + 7) / 8;   // 8 waves (rows) per block
  snn_prep<<<prep_blocks, 256, 0, stream>>>(x, xref, xh, rh, x2, r2, M, N, D);
  snn_main<<<M / 16, 256, 0, stream>>>(xh, rh, x2, r2, y, yref, lossv, N, D);
  snn_reduce<<<1, 256, 0, stream>>>(lossv, (float*)d_out, M);
}
